// ResBlock_DAQ_50852412785006
// MI455X (gfx1250) — 8-bit matrix operand order repaired and checked by reading, compiled, not run on hardware
//
#include <hip/hip_runtime.h>


typedef __attribute__((ext_vector_type(2))) int i2;
typedef __attribute__((ext_vector_type(4))) int i4;
typedef __attribute__((ext_vector_type(8))) int v8i;
typedef __attribute__((ext_vector_type(4))) unsigned uint32x4;
typedef __attribute__((ext_vector_type(4))) int int32x4;
typedef __attribute__((ext_vector_type(8))) int int32x8;

#define HW 128
#define CH 64
#define NB 16
#define NPIX (NB * HW * HW)
#define NELEM (NPIX * CH)
#define NW 36864
#define ROWB (HW * CH)
#define PROW (130 * CH)

#if defined(__gfx1250__) && __has_builtin(__builtin_amdgcn_tensor_load_to_lds) && \
    __has_builtin(__builtin_amdgcn_s_wait_tensorcnt)
#define HAS_TDM 1
#else
#define HAS_TDM 0
#endif

__global__ void zero_scalars_k(unsigned* s) { s[0] = 0u; s[1] = 0u; }

__global__ __launch_bounds__(256) void absmax_k(const float* __restrict__ x,
                                                unsigned n, unsigned* out) {
    float m = 0.f;
    for (unsigned i = blockIdx.x * 256u + threadIdx.x; i < n; i += gridDim.x * 256u)
        m = fmaxf(m, fabsf(x[i]));
    __shared__ float red[256];
    red[threadIdx.x] = m; __syncthreads();
    for (int s = 128; s > 0; s >>= 1) {
        if ((int)threadIdx.x < s) red[threadIdx.x] = fmaxf(red[threadIdx.x], red[threadIdx.x + s]);
        __syncthreads();
    }
    if (threadIdx.x == 0) atomicMax(out, __float_as_uint(red[0]));
}

__global__ __launch_bounds__(256) void wquant_pack_k(const float* __restrict__ w1,
                                                     const float* __restrict__ w2,
                                                     unsigned* __restrict__ qw1,
                                                     unsigned* __restrict__ qw2) {
    const float* w = blockIdx.x ? w2 : w1;
    unsigned* qw = blockIdx.x ? qw2 : qw1;

    float m = 0.f;
    for (int i = threadIdx.x; i < NW; i += 256) m = fmaxf(m, fabsf(tanhf(w[i])));
    __shared__ float red[256];
    red[threadIdx.x] = m; __syncthreads();
    for (int s = 128; s > 0; s >>= 1) {
        if ((int)threadIdx.x < s) red[threadIdx.x] = fmaxf(red[threadIdx.x], red[threadIdx.x + s]);
        __syncthreads();
    }
    const float inv = 1.f / (red[0] + 1e-8f);

    for (int d = threadIdx.x; d < 9216; d += 256) {
        const int v  = d & 7;
        const int L  = (d >> 3) & 31;
        const int n0 = (d >> 8) & 3;
        const int t  = d >> 10;
        const int n  = n0 * 16 + (L & 15);
        const int kb = ((v >> 2) * 32) + ((L >> 4) * 16) + ((v & 3) * 4);
        unsigned dw = 0u;
        for (int j = 0; j < 4; ++j) {
            const int cin = kb + j;
            const float wt = tanhf(w[(n * CH + cin) * 9 + t]);
            float q = rintf((wt * inv + 1.0f) * 1.5f);
            q = fminf(fmaxf(q, 0.f), 3.f);
            const int w3 = (int)(2.f * q) - 3;
            dw |= ((unsigned)(w3 & 0xFF)) << (8 * j);
        }
        qw[d] = dw;
    }
}

__global__ __launch_bounds__(256) void quant_x_k(const float* __restrict__ x,
                                                 const unsigned* __restrict__ s1b,
                                                 unsigned* __restrict__ qx) {
    const unsigned j = blockIdx.x * 256u + threadIdx.x;
    const float invs = 1.f / (__uint_as_float(*s1b) + 1e-8f);
    const int c4 = j & 15;
    const unsigned p = j >> 4;
    const int wp = p & 127, hp = (p >> 7) & 127, bp = p >> 14;
    unsigned dw = 0u;
    for (int k = 0; k < 4; ++k) {
        const int c = c4 * 4 + k;
        const float xv = x[(((bp * CH + c) * HW + hp) * HW) + wp];
        const int qi = (int)rintf(fminf(fmaxf(xv * invs, -1.f), 1.f));
        dw |= ((unsigned)(qi & 0xFF)) << (8 * k);
    }
    qx[j] = dw;
}

__device__ __forceinline__ void stage3rows(signed char* slds,
                                           const signed char* __restrict__ q,
                                           int bp, int hp) {
    const i4 z4 = {0, 0, 0, 0};
    if (hp == 0)
        for (int i = threadIdx.x; i < PROW / 16; i += 256) ((i4*)slds)[i] = z4;
    if (hp == HW - 1)
        for (int i = threadIdx.x; i < PROW / 16; i += 256) ((i4*)(slds + 2 * PROW))[i] = z4;
    if (threadIdx.x < 24) {
        const int r = threadIdx.x >> 3;
        const int side = (threadIdx.x >> 2) & 1;
        const int q16 = threadIdx.x & 3;
        ((i4*)(slds + r * PROW + side * (129 * CH)))[q16] = z4;
    }

    const int r0 = (hp == 0) ? 1 : 0;
    const int r1 = (hp == HW - 1) ? 1 : 2;

#if HAS_TDM
    if (threadIdx.x < 32) {
        for (int r = r0; r <= r1; ++r) {
            const unsigned lds_base =
                (unsigned)(unsigned long long)(slds + r * PROW + CH);
            const unsigned long long ga =
                (unsigned long long)(q + (((long)bp * HW) + (hp - 1 + r)) * (long)ROWB);
            uint32x4 g0 = {0u, 0u, 0u, 0u};
            g0[0] = 1u;
            g0[1] = lds_base;
            g0[2] = (unsigned)(ga & 0xFFFFFFFFull);
            g0[3] = (unsigned)((ga >> 32) & 0x01FFFFFFull) | (2u << 30);
            int32x8 g1 = {0, 0, 0, 0, 0, 0, 0, 0};
            g1[1] = (int)((ROWB & 0xFFFF) << 16);
            g1[2] = (int)((ROWB >> 16) | (1u << 16));
            g1[3] = (int)((ROWB & 0xFFFF) << 16);
            g1[4] = 1;
            g1[5] = ROWB;
            int32x4 gz4 = {0, 0, 0, 0};
            int32x8 gz8 = {0, 0, 0, 0, 0, 0, 0, 0};
            __builtin_amdgcn_tensor_load_to_lds(g0, g1, gz4, gz4, gz8, 0);
        }
        __builtin_amdgcn_s_wait_tensorcnt(0);
    }
#else
    for (int r = r0; r <= r1; ++r) {
        i4* dst = (i4*)(slds + r * PROW + CH);
        const i4* src = (const i4*)(q + (((long)bp * HW) + (hp - 1 + r)) * (long)ROWB);
        for (int i = threadIdx.x; i < ROWB / 16; i += 256) dst[i] = src[i];
    }
#endif
    __syncthreads();
}

__device__ __forceinline__ void conv_core(const signed char* slds,
                                          const v8i* __restrict__ qwv,
                                          int wbase, int lane, v8i acc[4]) {
    const int half = lane >> 4;
    const int mrw = wbase + (lane & 15);
    union AU { v8i v; i2 q[4]; };
    AU A, An;
    v8i bf[4], bn[4];

    {
        const signed char* rp = slds + mrw * CH;
#pragma unroll
        for (int k = 0; k < 4; ++k) A.q[k] = *(const i2*)(rp + half * 16 + (k >> 1) * 32 + (k & 1) * 8);
#pragma unroll
        for (int n0 = 0; n0 < 4; ++n0) bf[n0] = qwv[n0 * 32 + lane];
    }
#pragma unroll
    for (int t = 0; t < 9; ++t) {
        if (t < 8) {
            const int tn = t + 1;
            const signed char* rp = slds + (tn / 3) * PROW + (mrw + tn % 3) * CH;
#pragma unroll
            for (int k = 0; k < 4; ++k) An.q[k] = *(const i2*)(rp + half * 16 + (k >> 1) * 32 + (k & 1) * 8);
#pragma unroll
            for (int n0 = 0; n0 < 4; ++n0) bn[n0] = qwv[(tn * 4 + n0) * 32 + lane];
        }
#pragma unroll
        for (int n0 = 0; n0 < 4; ++n0)
            acc[n0] = __builtin_amdgcn_wmma_i32_16x16x64_iu8(
                true, A.v, true, bf[n0], acc[n0], false, false);
        if (t < 8) {
            A.v = An.v;
#pragma unroll
            for (int n0 = 0; n0 < 4; ++n0) bf[n0] = bn[n0];
        }
    }
}

__global__ __launch_bounds__(256) void conv1_k(const signed char* __restrict__ qx,
                                               const v8i* __restrict__ qwv,
                                               const float* __restrict__ b1,
                                               const unsigned* __restrict__ s1b,
                                               short* __restrict__ qacc,
                                               unsigned* __restrict__ s2b) {
    __shared__ signed char slds[3 * PROW];
    __shared__ float sb[CH];
    __shared__ float red[256];
    if (threadIdx.x < CH) sb[threadIdx.x] = b1[threadIdx.x];

    const int row = blockIdx.x;
    const int hp = row & 127, bp = row >> 7;
    stage3rows(slds, qx, bp, hp);

    const int lane = threadIdx.x & 31;
    const int half = lane >> 4;
    const int mr = lane & 15;
    const int wbase = (threadIdx.x >> 5) * 16;

    const v8i vz = {0, 0, 0, 0, 0, 0, 0, 0};
    v8i acc[4] = {vz, vz, vz, vz};
    conv_core(slds, qwv, wbase, lane, acc);

    const float mult = (__uint_as_float(*s1b) + 1e-8f) * (1.f / 3.f);
    float lmax = 0.f;
    const long pbase = ((long)bp * HW + hp) * HW + wbase;
#pragma unroll
    for (int n0 = 0; n0 < 4; ++n0) {
        const int N = n0 * 16 + mr;
#pragma unroll
        for (int r = 0; r < 8; ++r) {
            const int M = r + 8 * half;
            const int a = acc[n0][r];
            qacc[(pbase + M) * CH + N] = (short)a;
            lmax = fmaxf(lmax, fmaxf(mult * (float)a + sb[N], 0.f));
        }
    }
    red[threadIdx.x] = lmax; __syncthreads();
    for (int s = 128; s > 0; s >>= 1) {
        if ((int)threadIdx.x < s) red[threadIdx.x] = fmaxf(red[threadIdx.x], red[threadIdx.x + s]);
        __syncthreads();
    }
    if (threadIdx.x == 0) atomicMax(s2b, __float_as_uint(red[0]));
}

__global__ __launch_bounds__(256) void requant_k(const short* __restrict__ qacc,
                                                 const float* __restrict__ b1,
                                                 const unsigned* __restrict__ s1b,
                                                 const unsigned* __restrict__ s2b,
                                                 unsigned* __restrict__ qy) {
    const unsigned j = blockIdx.x * 256u + threadIdx.x;
    const float m1 = (__uint_as_float(*s1b) + 1e-8f) * (1.f / 3.f);
    const float inv2 = 1.f / (__uint_as_float(*s2b) + 1e-8f);
    const int c4 = j & 15;
    union { i2 v; short s[4]; } ch;
    ch.v = *(const i2*)(qacc + (size_t)j * 4);
    unsigned dw = 0u;
    for (int k = 0; k < 4; ++k) {
        const float y = fmaxf(m1 * (float)ch.s[k] + b1[c4 * 4 + k], 0.f);
        const int qi = (int)rintf(fminf(y * inv2, 1.f));
        dw |= ((unsigned)(qi & 0xFF)) << (8 * k);
    }
    qy[j] = dw;
}

__global__ __launch_bounds__(256) void conv2_k(const signed char* __restrict__ qy,
                                               const v8i* __restrict__ qwv,
                                               const float* __restrict__ b2,
                                               const float* __restrict__ x,
                                               const unsigned* __restrict__ s2b,
                                               float* __restrict__ out) {
    __shared__ signed char slds[3 * PROW];
    __shared__ float sb2[CH];
    if (threadIdx.x < CH) sb2[threadIdx.x] = b2[threadIdx.x];

    const int row = blockIdx.x;
    const int hp = row & 127, bp = row >> 7;
    stage3rows(slds, qy, bp, hp);

    const int lane = threadIdx.x & 31;
    const int half = lane >> 4;
    const int mr = lane & 15;
    const int wbase = (threadIdx.x >> 5) * 16;

    const v8i vz = {0, 0, 0, 0, 0, 0, 0, 0};
    v8i acc[4] = {vz, vz, vz, vz};
    conv_core(slds, qwv, wbase, lane, acc);

    const float m2 = (__uint_as_float(*s2b) + 1e-8f) * (1.f / 3.f);
#pragma unroll
    for (int n0 = 0; n0 < 4; ++n0) {
        const int N = n0 * 16 + mr;
#pragma unroll
        for (int r = 0; r < 8; ++r) {
            const int M = r + 8 * half;
            const long oidx = (((long)bp * CH + N) * HW + hp) * HW + (wbase + M);
            out[oidx] = m2 * (float)acc[n0][r] + sb2[N] + x[oidx];
        }
    }
}

extern "C" void kernel_launch(void* const* d_in, const int* in_sizes, int n_in,
                              void* d_out, int out_size, void* d_ws, size_t ws_size,
                              hipStream_t stream) {
    (void)in_sizes; (void)n_in; (void)out_size; (void)ws_size;
    const float* x  = (const float*)d_in[0];
    const float* w1 = (const float*)d_in[1];
    const float* b1 = (const float*)d_in[2];
    const float* w2 = (const float*)d_in[3];
    const float* b2 = (const float*)d_in[4];
    float* out = (float*)d_out;

    char* ws = (char*)d_ws;
    unsigned* scal = (unsigned*)ws;
    unsigned* qw1  = (unsigned*)(ws + 128);
    unsigned* qw2  = (unsigned*)(ws + 128 + NW);
    unsigned* qx   = (unsigned*)(ws + 128 + 2 * NW);
    short*    qacc = (short*)(ws + 128 + 2 * NW + (size_t)NELEM);
    unsigned* qy   = (unsigned*)(ws + 128 + 2 * NW + 3 * (size_t)NELEM);

    zero_scalars_k<<<1, 1, 0, stream>>>(scal);
    absmax_k<<<4096, 256, 0, stream>>>(x, (unsigned)NELEM, scal + 0);
    wquant_pack_k<<<2, 256, 0, stream>>>(w1, w2, qw1, qw2);
    quant_x_k<<<NELEM / 4 / 256, 256, 0, stream>>>(x, scal + 0, qx);
    conv1_k<<<NB * HW, 256, 0, stream>>>((const signed char*)qx, (const v8i*)qw1,
                                         b1, scal + 0, qacc, scal + 1);
    requant_k<<<NELEM / 4 / 256, 256, 0, stream>>>(qacc, b1, scal + 0, scal + 1, qy);
    conv2_k<<<NB * HW, 256, 0, stream>>>((const signed char*)qy, (const v8i*)qw2,
                                         b2, x, scal + 1, out);
}
